// cross_layer_6906307411983
// MI455X (gfx1250) — matrix operand order repaired and checked by reading, not compiled, not run on hardware
//
#include <hip/hip_runtime.h>

typedef __attribute__((ext_vector_type(16))) __bf16 v16bf;
typedef __attribute__((ext_vector_type(8)))  float  v8f;
typedef __attribute__((ext_vector_type(4)))  unsigned int u32x4;
typedef __attribute__((ext_vector_type(8)))  int    i32x8;
typedef __attribute__((ext_vector_type(4)))  int    i32x4;

#define E_LAT  256
#define MDIM   512
#define DDIM   256
#define NHEADS 8
#define DH     64
#define FF     2048
#define TN     32
#define NCHUNK 32

union V16 {
  v16bf v;
  __bf16 b[16];
  unsigned short h[16];
  u32x4 q[2];
};

__device__ __forceinline__ unsigned short bf16s(float f) {
  return __builtin_bit_cast(unsigned short, (__bf16)f);   // v_cvt to bf16
}

__device__ __forceinline__ v8f wmma_bf16(v16bf a, v16bf b, v8f c) {
  return __builtin_amdgcn_wmma_f32_16x16x32_bf16(false, a, false, b, (short)0, c, false, false);
}

// A tile (16x32 bf16) from bf16 row-major; lane layout => two contiguous 16B chunks
__device__ __forceinline__ v16bf lda_u16(const unsigned short* s, int row0, int k0, int ld, int lane) {
  int m = lane & 15, hh = lane >> 4;
  const unsigned short* p = s + (size_t)(row0 + m) * ld + k0 + 8 * hh;
  V16 a;
  a.q[0] = *(const u32x4*)p;          // K = 8*hh .. 8*hh+7
  a.q[1] = *(const u32x4*)(p + 16);   // K = 16+8*hh .. +7
  return a.v;
}
// A tile from fp32 row-major with on-the-fly bf16 convert (vectorizable 32B chunks)
__device__ __forceinline__ v16bf lda_f32(const float* s, int row0, int k0, int ld, int lane) {
  int m = lane & 15, hh = lane >> 4;
  const float* p = s + (size_t)(row0 + m) * ld + k0 + 8 * hh;
  V16 a;
#pragma unroll
  for (int i = 0; i < 8; ++i) a.b[i] = (__bf16)p[i];
#pragma unroll
  for (int i = 0; i < 8; ++i) a.b[8 + i] = (__bf16)p[16 + i];
  return a.v;
}
// B tile (32x16), K-contiguous bf16; caller pre-offsets to its N column's K run
__device__ __forceinline__ v16bf ldb_u16(const unsigned short* s, int lane) {
  const unsigned short* p = s + (lane >> 4) * 8;  // lanes 0-15: K 0..7,16..23; 16-31: K 8..15,24..31
  V16 b;
  b.q[0] = *(const u32x4*)p;
  b.q[1] = *(const u32x4*)(p + 16);
  return b.v;
}

// C = A(fp32, 16xK) * Wbf^T tile: B(k,n) = Wbf[(col0+n)*ldw + k]  (K-contiguous)
__device__ __forceinline__ v8f gemm_tile_ABt(const float* A, int lda, const unsigned short* Wbf,
                                             int ldw, int row0, int col0, int K, int lane) {
  v8f c = {};
  int ln = lane & 15;
  for (int k0 = 0; k0 < K; k0 += 32) {
    v16bf a = lda_f32(A, row0, k0, lda, lane);
    v16bf b = ldb_u16(Wbf + (size_t)(col0 + ln) * ldw + k0, lane);
    c = wmma_bf16(a, b, c);
  }
  return c;
}

// ---------------- fp32 -> bf16 conversion (x and weights, once) ----------------
__global__ __launch_bounds__(256) void cvt_bf16_kernel(const float* __restrict__ src,
                                                       unsigned short* __restrict__ dst, int count) {
  int i = blockIdx.x * 256 + threadIdx.x;
  int stride = gridDim.x * 256;
  for (; i < count; i += stride) dst[i] = bf16s(src[i]);
}

// ---------------- Q projection: Q = M @ Wq^T -> bf16 [h][e][dh] ----------------
__global__ __launch_bounds__(256) void qproj_kernel(const float* __restrict__ Mlat,
                                                    const unsigned short* __restrict__ Wqbf,
                                                    unsigned short* __restrict__ Qbf) {
  int tid = threadIdx.x, wid = tid >> 5, lane = tid & 31;
  int ln = lane & 15, hh = lane >> 4;
  int id = blockIdx.x * 8 + wid;          // 512 tiles: 16 rb x 32 cb
  int rb = id >> 5, cb = id & 31;
  v8f c = gemm_tile_ABt(Mlat, MDIM, Wqbf, MDIM, rb * 16, cb * 16, MDIM, lane);
  int col = cb * 16 + ln;
  int hcol = col >> 6, dh = col & 63;
#pragma unroll
  for (int r = 0; r < 8; ++r) {
    int row = rb * 16 + r + 8 * hh;
    Qbf[(hcol * E_LAT + row) * DH + dh] = bf16s(c[r]);
  }
}

// ---------------- TDM: DMA one 32x256 bf16 x-tile into LDS ----------------
// D# group0/1 per CDNA5 ISA 8.3/8.4: count=1, type=2, data_size=2B,
// tensor = (dim0=256, dim1=n, stride0=256), tile = (256, TN).
__device__ __forceinline__ void tdm_load_xtile(const unsigned short* xbf, unsigned lds_off,
                                               int nbase, int n) {
  unsigned long long ga = (unsigned long long)xbf + (unsigned long long)nbase * (DDIM * 2);
  u32x4 g0;
  g0[0] = 1u;                                   // count=1 valid descriptor
  g0[1] = lds_off;                              // lds_addr (bytes)
  g0[2] = (unsigned)ga;                         // global_addr[31:0]
  g0[3] = (unsigned)((ga >> 32) & 0x1ffffffull) | (2u << 30);  // addr[56:32] | type=2
  i32x8 g1;
  g1[0] = (int)(1u << 16);                      // data_size=1 (2 bytes)
  g1[1] = (int)(((unsigned)DDIM & 0xffffu) << 16);             // tensor_dim0[15:0]
  g1[2] = (int)((((unsigned)DDIM >> 16) & 0xffffu) | (((unsigned)n & 0xffffu) << 16));
  g1[3] = (int)((((unsigned)n >> 16) & 0xffffu) | ((unsigned)DDIM << 16)); // tile_dim0=256
  g1[4] = TN;                                   // tile_dim1=32, tile_dim2=0
  g1[5] = DDIM;                                 // tensor_dim0_stride[31:0]
  g1[6] = 0;
  g1[7] = 0;
  i32x4 z4 = {0, 0, 0, 0};
  i32x8 z8 = {0, 0, 0, 0, 0, 0, 0, 0};
  __builtin_amdgcn_tensor_load_to_lds(g0, g1, z4, z4, z8, 0);  // 6-arg (clang-23) form
}

// ---------------- Flash attention (split over n-chunks, TDM double-buffered) ----------------
__global__ __launch_bounds__(256) void attn_kernel(
    const unsigned short* __restrict__ xbf, const unsigned char* __restrict__ mask,
    const unsigned short* __restrict__ Wkvbf, const float* __restrict__ bkv,
    const unsigned short* __restrict__ Qbf,
    float* __restrict__ Opart, float* __restrict__ mpart, float* __restrict__ lpart, int n) {
  __shared__ unsigned short xs[2][TN * DDIM];  // x tiles bf16, double buffer  2x16KB
  __shared__ unsigned short ks[TN * DH];       // k bf16 [n][dh]  (K-contig for QK^T) 4KB
  __shared__ unsigned short vt[DH * TN];       // v bf16 [dh][n]  (K-contig for PV)   4KB
  __shared__ unsigned short ps[E_LAT * TN];    // P bf16 [256][TN]                   16KB

  const int h = blockIdx.x, chunk = blockIdx.y;
  const int tid = threadIdx.x, wid = tid >> 5, lane = tid & 31;
  const int ln = lane & 15, hh = lane >> 4;

  const int ntiles = (n + TN - 1) / TN;
  const int tpc = (ntiles + NCHUNK - 1) / NCHUNK;
  const int t0 = chunk * tpc;
  const int t1 = (t0 + tpc < ntiles) ? (t0 + tpc) : ntiles;

  const unsigned short* Qh = Qbf + h * (E_LAT * DH);
  const float scale = 0.125f;  // 64^-0.5

  v8f zero = {};
  v8f acc[2][4];
  float mrow[2][8], lrow[2][8];
#pragma unroll
  for (int a = 0; a < 2; ++a) {
#pragma unroll
    for (int cb = 0; cb < 4; ++cb) acc[a][cb] = zero;
#pragma unroll
    for (int r = 0; r < 8; ++r) { mrow[a][r] = -3.0e38f; lrow[a][r] = 0.f; }
  }

  // prologue: issue DMA for first tile (wave 0 only; TDM is a wave-level op)
  if (wid == 0 && t0 < t1)
    tdm_load_xtile(xbf, (unsigned)(unsigned long long)&xs[0][0], t0 * TN, n);

  for (int t = t0; t < t1; ++t) {
    const int cur = (t - t0) & 1;
    const int nbase = t * TN;
    if (wid == 0) __builtin_amdgcn_s_wait_tensorcnt(0);
    __syncthreads();  // xs[cur] visible to all waves
    if (wid == 0 && t + 1 < t1)
      tdm_load_xtile(xbf, (unsigned)(unsigned long long)&xs[cur ^ 1][0], (t + 1) * TN, n);
    const unsigned short* xc = &xs[cur][0];

    // ---- kv projection: 16 output tiles (k:8, v:8), 2 per wave ----
#pragma unroll
    for (int s = 0; s < 2; ++s) {
      int id = wid * 2 + s;
      int kv = id >> 3, rb = (id >> 2) & 1, cb = id & 3;
      int colbase = kv * 512 + h * 64 + cb * 16;
      v8f vacc = zero;
#pragma unroll
      for (int kk = 0; kk < 8; ++kk) {
        v16bf a = lda_u16(xc, rb * 16, kk * 32, DDIM, lane);
        v16bf b = ldb_u16(Wkvbf + (size_t)(colbase + ln) * DDIM + kk * 32, lane);
        vacc = wmma_bf16(a, b, vacc);
      }
      float bv = bkv[colbase + ln];
#pragma unroll
      for (int r = 0; r < 8; ++r) {
        float val = vacc[r] + bv;
        int nl = rb * 16 + r + 8 * hh;
        if (kv == 0) ks[nl * DH + cb * 16 + ln] = bf16s(val);        // [n][dh]
        else         vt[(cb * 16 + ln) * TN + nl] = bf16s(val);      // [dh][n]
      }
    }
    __syncthreads();

    // ---- scores for this wave's 32-row band ----
    v8f sc[2][2];
#pragma unroll
    for (int a = 0; a < 2; ++a)
#pragma unroll
      for (int cb = 0; cb < 2; ++cb) {
        v8f s = zero;
#pragma unroll
        for (int kk = 0; kk < 2; ++kk) {
          v16bf av = lda_u16(Qh, 32 * wid + a * 16, kk * 32, DH, lane);
          v16bf bv = ldb_u16(ks + (cb * 16 + ln) * DH + kk * 32, lane);
          s = wmma_bf16(av, bv, s);
        }
        sc[a][cb] = s;
      }

    // ---- online softmax (wave-local rows, shuffle reductions in 16-lane halves) ----
#pragma unroll
    for (int a = 0; a < 2; ++a) {
#pragma unroll
      for (int r = 0; r < 8; ++r) {
        int row = 32 * wid + a * 16 + r + 8 * hh;
        float v0 = sc[a][0][r] * scale;
        float v1 = sc[a][1][r] * scale;
        int j0 = nbase + ln, j1 = nbase + 16 + ln;
        bool k0 = (j0 < n) && mask[(size_t)row * n + j0];
        bool k1 = (j1 < n) && mask[(size_t)row * n + j1];
        v0 = k0 ? v0 : -1e9f;
        v1 = k1 ? v1 : -1e9f;
        float rm = fmaxf(v0, v1);
#pragma unroll
        for (int mk = 8; mk >= 1; mk >>= 1) rm = fmaxf(rm, __shfl_xor(rm, mk, 32));
        float mnew = fmaxf(mrow[a][r], rm);
        float corr = __expf(mrow[a][r] - mnew);
        float p0 = __expf(v0 - mnew), p1 = __expf(v1 - mnew);
        float lsum = p0 + p1;
#pragma unroll
        for (int mk = 8; mk >= 1; mk >>= 1) lsum += __shfl_xor(lsum, mk, 32);
        lrow[a][r] = lrow[a][r] * corr + lsum;
        mrow[a][r] = mnew;
#pragma unroll
        for (int cb = 0; cb < 4; ++cb) acc[a][cb][r] *= corr;
        ps[row * TN + ln] = bf16s(p0);
        ps[row * TN + 16 + ln] = bf16s(p1);
      }
    }

    // ---- O += P @ V (K = TN = 32, single WMMA per tile) ----
#pragma unroll
    for (int a = 0; a < 2; ++a) {
      v16bf pa = lda_u16(ps, 32 * wid + a * 16, 0, TN, lane);
#pragma unroll
      for (int cb = 0; cb < 4; ++cb) {
        v16bf bv = ldb_u16(vt + (cb * 16 + ln) * TN, lane);
        acc[a][cb] = wmma_bf16(pa, bv, acc[a][cb]);
      }
    }
    __syncthreads();  // ks/vt/ps consumed before next iteration rewrites them
  }

  // ---- write partials ----
  float* Ob = Opart + (size_t)((h * NCHUNK + chunk) * E_LAT) * DH;
#pragma unroll
  for (int a = 0; a < 2; ++a)
#pragma unroll
    for (int cb = 0; cb < 4; ++cb)
#pragma unroll
      for (int r = 0; r < 8; ++r) {
        int row = 32 * wid + a * 16 + r + 8 * hh;
        Ob[row * DH + cb * 16 + ln] = acc[a][cb][r];
      }
  if (ln == 0) {  // lanes 0 and 16 own distinct rows
#pragma unroll
    for (int a = 0; a < 2; ++a)
#pragma unroll
      for (int r = 0; r < 8; ++r) {
        int row = 32 * wid + a * 16 + r + 8 * hh;
        mpart[(h * NCHUNK + chunk) * E_LAT + row] = mrow[a][r];
        lpart[(h * NCHUNK + chunk) * E_LAT + row] = lrow[a][r];
      }
  }
}

// ---------------- combine split-K partials (log-sum-exp merge) ----------------
__global__ void combine_kernel(const float* __restrict__ Opart, const float* __restrict__ mpart,
                               const float* __restrict__ lpart, float* __restrict__ Aout) {
  int idx = blockIdx.x;           // h*256 + e
  int h = idx >> 8, e = idx & 255;
  int j = threadIdx.x;            // 0..63
  float Mg = -3.0e38f;
  for (int c = 0; c < NCHUNK; ++c) Mg = fmaxf(Mg, mpart[(h * NCHUNK + c) * E_LAT + e]);
  float num = 0.f, den = 0.f;
  for (int c = 0; c < NCHUNK; ++c) {
    int b = (h * NCHUNK + c) * E_LAT + e;
    float w = __expf(mpart[b] - Mg);
    den += w * lpart[b];
    num += w * Opart[(size_t)b * DH + j];
  }
  Aout[e * MDIM + h * DH + j] = num / fmaxf(den, 1e-30f);
}

// ---------------- out-proj: T1 = A @ Wo^T + bo + M ----------------
__global__ __launch_bounds__(256) void woproj_kernel(const float* __restrict__ A,
                                                     const unsigned short* __restrict__ Wobf,
                                                     const float* __restrict__ bo,
                                                     const float* __restrict__ Mlat,
                                                     float* __restrict__ T1) {
  int tid = threadIdx.x, wid = tid >> 5, lane = tid & 31;
  int ln = lane & 15, hh = lane >> 4;
  int id = blockIdx.x * 8 + wid;
  int rb = id >> 5, cb = id & 31;
  v8f c = gemm_tile_ABt(A, MDIM, Wobf, MDIM, rb * 16, cb * 16, MDIM, lane);
  int col = cb * 16 + ln;
#pragma unroll
  for (int r = 0; r < 8; ++r) {
    int row = rb * 16 + r + 8 * hh;
    T1[row * MDIM + col] = c[r] + bo[col] + Mlat[row * MDIM + col];
  }
}

// ---------------- layer norm (no affine), one wave per 512-wide row ----------------
__global__ __launch_bounds__(256) void ln_kernel(const float* __restrict__ in,
                                                 float* __restrict__ out) {
  int tid = threadIdx.x, wid = tid >> 5, lane = tid & 31;
  int row = blockIdx.x * 8 + wid;
  const float* p = in + row * MDIM;
  float s = 0.f, sq = 0.f;
  for (int i = lane; i < MDIM; i += 32) { float v = p[i]; s += v; sq += v * v; }
#pragma unroll
  for (int mk = 16; mk >= 1; mk >>= 1) { s += __shfl_xor(s, mk, 32); sq += __shfl_xor(sq, mk, 32); }
  float mu = s * (1.f / MDIM);
  float var = sq * (1.f / MDIM) - mu * mu;
  float rs = rsqrtf(var + 1e-5f);
  for (int i = lane; i < MDIM; i += 32) out[row * MDIM + i] = (p[i] - mu) * rs;
}

// ---------------- FFN1: H = relu(m @ W1^T + b1) ----------------
__global__ __launch_bounds__(256) void ffn1_kernel(const float* __restrict__ mbuf,
                                                   const unsigned short* __restrict__ W1bf,
                                                   const float* __restrict__ b1,
                                                   float* __restrict__ H) {
  int tid = threadIdx.x, wid = tid >> 5, lane = tid & 31;
  int ln = lane & 15, hh = lane >> 4;
  int id = blockIdx.x * 8 + wid;         // 2048 tiles: 16 rb x 128 cb
  int rb = id >> 7, cb = id & 127;
  v8f c = gemm_tile_ABt(mbuf, MDIM, W1bf, MDIM, rb * 16, cb * 16, MDIM, lane);
  int col = cb * 16 + ln;
#pragma unroll
  for (int r = 0; r < 8; ++r) {
    int row = rb * 16 + r + 8 * hh;
    H[row * FF + col] = fmaxf(c[r] + b1[col], 0.f);
  }
}

// ---------------- FFN2: T2 = H @ W2^T + b2 + m ----------------
__global__ __launch_bounds__(256) void ffn2_kernel(const float* __restrict__ H,
                                                   const unsigned short* __restrict__ W2bf,
                                                   const float* __restrict__ b2,
                                                   const float* __restrict__ mbuf,
                                                   float* __restrict__ T2) {
  int tid = threadIdx.x, wid = tid >> 5, lane = tid & 31;
  int ln = lane & 15, hh = lane >> 4;
  int id = blockIdx.x * 8 + wid;
  int rb = id >> 5, cb = id & 31;
  v8f c = gemm_tile_ABt(H, FF, W2bf, FF, rb * 16, cb * 16, FF, lane);
  int col = cb * 16 + ln;
#pragma unroll
  for (int r = 0; r < 8; ++r) {
    int row = rb * 16 + r + 8 * hh;
    T2[row * MDIM + col] = c[r] + b2[col] + mbuf[row * MDIM + col];
  }
}

extern "C" void kernel_launch(void* const* d_in, const int* in_sizes, int n_in,
                              void* d_out, int out_size, void* d_ws, size_t ws_size,
                              hipStream_t stream) {
  (void)n_in; (void)out_size; (void)ws_size;
  const float* Mlat = (const float*)d_in[0];
  const float* x    = (const float*)d_in[1];
  const unsigned char* mask = (const unsigned char*)d_in[2];
  const float* Wq   = (const float*)d_in[3];
  const float* Wkv  = (const float*)d_in[4];
  const float* bkv  = (const float*)d_in[5];
  const float* Wo   = (const float*)d_in[6];
  const float* bo   = (const float*)d_in[7];
  const float* W1   = (const float*)d_in[8];
  const float* b1   = (const float*)d_in[9];
  const float* W2   = (const float*)d_in[10];
  const float* b2   = (const float*)d_in[11];
  const int n = in_sizes[1] / DDIM;

  char* w = (char*)d_ws;
  unsigned short* Qbf  = (unsigned short*)w; w += (size_t)NHEADS * E_LAT * DH * 2;
  unsigned short* xbf  = (unsigned short*)w; w += (size_t)n * DDIM * 2;
  unsigned short* Wqbf = (unsigned short*)w; w += (size_t)MDIM * MDIM * 2;
  unsigned short* Wkvbf= (unsigned short*)w; w += (size_t)2 * MDIM * DDIM * 2;
  unsigned short* Wobf = (unsigned short*)w; w += (size_t)MDIM * MDIM * 2;
  unsigned short* W1bf = (unsigned short*)w; w += (size_t)FF * MDIM * 2;
  unsigned short* W2bf = (unsigned short*)w; w += (size_t)MDIM * FF * 2;
  float* Opart = (float*)w; w += (size_t)NHEADS * NCHUNK * E_LAT * DH * 4;
  float* mpart = (float*)w; w += (size_t)NHEADS * NCHUNK * E_LAT * 4;
  float* lpart = (float*)w; w += (size_t)NHEADS * NCHUNK * E_LAT * 4;
  float* Aout  = (float*)w; w += (size_t)E_LAT * MDIM * 4;
  float* T1    = (float*)w; w += (size_t)E_LAT * MDIM * 4;
  float* mbuf  = (float*)w; w += (size_t)E_LAT * MDIM * 4;
  float* Hbuf  = (float*)w; w += (size_t)E_LAT * FF * 4;
  float* T2    = (float*)w; w += (size_t)E_LAT * MDIM * 4;

  cvt_bf16_kernel<<<2048, 256, 0, stream>>>(x, xbf, n * DDIM);
  cvt_bf16_kernel<<<256, 256, 0, stream>>>(Wq, Wqbf, MDIM * MDIM);
  cvt_bf16_kernel<<<256, 256, 0, stream>>>(Wkv, Wkvbf, 2 * MDIM * DDIM);
  cvt_bf16_kernel<<<256, 256, 0, stream>>>(Wo, Wobf, MDIM * MDIM);
  cvt_bf16_kernel<<<512, 256, 0, stream>>>(W1, W1bf, FF * MDIM);
  cvt_bf16_kernel<<<512, 256, 0, stream>>>(W2, W2bf, MDIM * FF);

  qproj_kernel<<<64, 256, 0, stream>>>(Mlat, Wqbf, Qbf);
  attn_kernel<<<dim3(NHEADS, NCHUNK), 256, 0, stream>>>(xbf, mask, Wkvbf, bkv, Qbf,
                                                        Opart, mpart, lpart, n);
  combine_kernel<<<NHEADS * E_LAT, 64, 0, stream>>>(Opart, mpart, lpart, Aout);
  woproj_kernel<<<64, 256, 0, stream>>>(Aout, Wobf, bo, Mlat, T1);
  ln_kernel<<<32, 256, 0, stream>>>(T1, mbuf);
  ffn1_kernel<<<256, 256, 0, stream>>>(mbuf, W1bf, b1, Hbuf);
  ffn2_kernel<<<64, 256, 0, stream>>>(Hbuf, W2bf, b2, mbuf, T2);
  ln_kernel<<<32, 256, 0, stream>>>(T2, (float*)d_out);
}
